// GraphGNN_35845797053146
// MI455X (gfx1250) — matrix operand order repaired and checked by reading, not compiled, not run on hardware
//
#include <hip/hip_runtime.h>
#include <hip/hip_bf16.h>

typedef __bf16 bf16_t;
typedef __attribute__((ext_vector_type(16))) __bf16 v16bf;
typedef __attribute__((ext_vector_type(8)))  float  v8f;
typedef __attribute__((ext_vector_type(4)))  int    v4i;

union FragBF { v16bf v; unsigned u[8]; };
union FragF  { v8f v; float f[8]; };

#define GN 50000
#define GE 800000
#define GD 512
#define GH 512
#define GG 64
#define GC 10

#if defined(__HIP_DEVICE_COMPILE__) && __has_builtin(__builtin_amdgcn_global_load_async_to_lds_b128)
#define HAS_ASYNC_LDS 1
#else
#define HAS_ASYNC_LDS 0
#endif

// 16-byte global -> LDS copy; async (CDNA5 GLOBAL_LOAD_ASYNC_TO_LDS_B128) when available.
// Builtin signature (probed via diagnostics): (int4 AS1* gsrc, int4 AS3* ldst, Ii off, Ii cpol)
__device__ __forceinline__ void cp16_g2lds(void* lds, const void* g) {
#if HAS_ASYNC_LDS
    __builtin_amdgcn_global_load_async_to_lds_b128(
        (__attribute__((address_space(1))) v4i*)g,
        (__attribute__((address_space(3))) v4i*)lds, 0, 0);
#else
    *(uint4*)lds = *(const uint4*)g;
#endif
}

__device__ __forceinline__ void wait_async_all() {
#if HAS_ASYNC_LDS
#if __has_builtin(__builtin_amdgcn_s_wait_asynccnt)
    __builtin_amdgcn_s_wait_asynccnt(0);
#else
    asm volatile("s_wait_asynccnt 0" ::: "memory");
#endif
#endif
}

// ---------------------------------------------------------------------------
// fp32 -> bf16 elementwise convert
__global__ void k_f2bf(const float* __restrict__ src, bf16_t* __restrict__ dst,
                       long long n) {
    long long i = (long long)blockIdx.x * blockDim.x + threadIdx.x;
    if (i < n) dst[i] = (bf16_t)src[i];
}

// fp32 weight [K][N] -> bf16 transposed [N][K]
__global__ void k_w2bft(const float* __restrict__ w, bf16_t* __restrict__ wt) {
    int i = blockIdx.x * blockDim.x + threadIdx.x;
    if (i >= GD * GH) return;
    int k = i / GH, n = i % GH;
    wt[(size_t)n * GD + k] = (bf16_t)w[(size_t)k * GH + n];
}

// ---------------------------------------------------------------------------
// Edge scatter-sum: one wave per edge (endpoint loads wave-uniform),
// each lane owns 16 contiguous features.
template <typename T>
__global__ __launch_bounds__(256) void k_aggregate(const T* __restrict__ feat,
                                                   const int* __restrict__ ei,
                                                   float* __restrict__ agg) {
    int e = blockIdx.x * 8 + (threadIdx.x >> 5);
    if (e >= GE) return;
    int lane = threadIdx.x & 31;
    int s = ei[e];
    int d = ei[GE + e];
    const T* sp = feat + (size_t)s * GD + lane * 16;
    float* dp = agg + (size_t)d * GD + lane * 16;
#pragma unroll
    for (int j = 0; j < 16; j++) atomicAdd(dp + j, (float)sp[j]);
}

__global__ void k_count(const int* __restrict__ batch, float* __restrict__ cnt) {
    int i = blockIdx.x * blockDim.x + threadIdx.x;
    if (i < GN) atomicAdd(&cnt[batch[i]], 1.0f);
}

// ---------------------------------------------------------------------------
// Fused GraphConv GEMM: out = relu(Aagg @ Wr + bias + Aself @ Ws)
//   block tile 128x128, 8 waves, wave tile 32x64 (2x4 WMMA 16x16 frags)
//   double-buffered LDS staging via async global->LDS copies
//   pool_mode: 0 -> store bf16 activations; 1 -> atomic mean-pool accumulate
__global__ __launch_bounds__(256) void k_gcn_gemm(
    const bf16_t* __restrict__ Aagg, const bf16_t* __restrict__ Aself,
    const bf16_t* __restrict__ BrT, const bf16_t* __restrict__ BsT,
    const float* __restrict__ bias, bf16_t* __restrict__ outBf,
    float* __restrict__ pooled, const int* __restrict__ batch,
    int nrows, int pool_mode) {
    __shared__ __align__(16) bf16_t sAa[2][128][32];
    __shared__ __align__(16) bf16_t sAs[2][128][32];
    __shared__ __align__(16) bf16_t sBr[2][128][32];
    __shared__ __align__(16) bf16_t sBs[2][128][32];

    const int tid = threadIdx.x;
    const int lane = tid & 31;
    const int wave = tid >> 5;
    const int lm = lane & 15;   // row/col within 16x16 fragment
    const int lh = lane >> 4;   // lane half

    const int mBase = blockIdx.y * 128;
    const int nBase = blockIdx.x * 128;
    const int waveM = (wave & 3) * 32;
    const int waveN = (wave >> 2) * 64;

    FragF acc[2][4];
#pragma unroll
    for (int i = 0; i < 2; i++)
#pragma unroll
        for (int j = 0; j < 4; j++)
#pragma unroll
            for (int r = 0; r < 8; r++) acc[i][j].f[r] = 0.f;

    const int sRow = tid >> 1;         // 128 rows, 2 threads per row
    const int sOff = (tid & 1) * 16;   // 16 bf16 (32 B) per thread
    const bool rowOk = (mBase + sRow) < nrows;

    // Stage one 32-wide K-slice of all four tiles into LDS buffer `buf`.
    auto stage = [&](int buf, int k0) {
        if (rowOk) {
            const bf16_t* pa = Aagg + (size_t)(mBase + sRow) * GD + k0 + sOff;
            const bf16_t* px = Aself + (size_t)(mBase + sRow) * GD + k0 + sOff;
            cp16_g2lds(&sAa[buf][sRow][sOff], pa);
            cp16_g2lds(&sAa[buf][sRow][sOff + 8], pa + 8);
            cp16_g2lds(&sAs[buf][sRow][sOff], px);
            cp16_g2lds(&sAs[buf][sRow][sOff + 8], px + 8);
        }
        const bf16_t* pr = BrT + (size_t)(nBase + sRow) * GD + k0 + sOff;
        const bf16_t* ps = BsT + (size_t)(nBase + sRow) * GD + k0 + sOff;
        cp16_g2lds(&sBr[buf][sRow][sOff], pr);
        cp16_g2lds(&sBr[buf][sRow][sOff + 8], pr + 8);
        cp16_g2lds(&sBs[buf][sRow][sOff], ps);
        cp16_g2lds(&sBs[buf][sRow][sOff + 8], ps + 8);
    };

    // Out-of-range A rows: zero once; async staging is predicated off for them,
    // so the zeros persist across all K-steps.
    if (!rowOk) {
        const uint4 z4 = make_uint4(0u, 0u, 0u, 0u);
#pragma unroll
        for (int b = 0; b < 2; b++) {
            *(uint4*)&sAa[b][sRow][sOff] = z4;
            *(uint4*)&sAa[b][sRow][sOff + 8] = z4;
            *(uint4*)&sAs[b][sRow][sOff] = z4;
            *(uint4*)&sAs[b][sRow][sOff + 8] = z4;
        }
    }

    stage(0, 0);
    wait_async_all();
    __syncthreads();

    for (int k0 = 0; k0 < GD; k0 += 32) {
        const int buf = (k0 >> 5) & 1;
        if (k0 + 32 < GD) stage(buf ^ 1, k0 + 32);

        // ---- build fragments (ISA 7.12.2 16-bit layouts) ----
        FragBF aA[2], aS[2], bR[4], bS[4];
#pragma unroll
        for (int mi = 0; mi < 2; mi++) {
            int row = waveM + mi * 16 + lm;
#pragma unroll
            for (int j = 0; j < 8; j++) {
                int kk = (j >> 2) * 16 + lh * 8 + (j & 3) * 2;  // A: 16x32 bf16
                aA[mi].u[j] = *(const unsigned*)&sAa[buf][row][kk];
                aS[mi].u[j] = *(const unsigned*)&sAs[buf][row][kk];
            }
        }
#pragma unroll
        for (int ni = 0; ni < 4; ni++) {
            int col = waveN + ni * 16 + lm;
#pragma unroll
            for (int j = 0; j < 8; j++) {
                int kk = (j >> 2) * 16 + lh * 8 + (j & 3) * 2;  // B: 32x16 bf16 (stored transposed)
                bR[ni].u[j] = *(const unsigned*)&sBr[buf][col][kk];
                bS[ni].u[j] = *(const unsigned*)&sBs[buf][col][kk];
            }
        }

        // ---- 16 WMMAs per K-step: agg@Wr and self@Ws share accumulators ----
#pragma unroll
        for (int mi = 0; mi < 2; mi++)
#pragma unroll
            for (int ni = 0; ni < 4; ni++) {
                acc[mi][ni].v = __builtin_amdgcn_wmma_f32_16x16x32_bf16(
                    false, aA[mi].v, false, bR[ni].v, (short)0, acc[mi][ni].v,
                    false, false);
                acc[mi][ni].v = __builtin_amdgcn_wmma_f32_16x16x32_bf16(
                    false, aS[mi].v, false, bS[ni].v, (short)0, acc[mi][ni].v,
                    false, false);
            }

        wait_async_all();   // next buffer's async copies have landed
        __syncthreads();    // ...and are visible to all waves
    }

    // ---- epilogue: bias + relu, then store bf16 or pool-accumulate ----
#pragma unroll
    for (int mi = 0; mi < 2; mi++) {
#pragma unroll
        for (int ni = 0; ni < 4; ni++) {
            int col = nBase + waveN + ni * 16 + lm;
            float b = bias[col];
#pragma unroll
            for (int r = 0; r < 8; r++) {
                int row = mBase + waveM + mi * 16 + lh * 8 + r;  // C-layout: VGPR r -> M=r / r+8
                if (row < nrows) {
                    float vv = acc[mi][ni].f[r] + b;
                    vv = vv > 0.f ? vv : 0.f;
                    if (pool_mode) {
                        atomicAdd(&pooled[(size_t)batch[row] * GH + col], vv);
                    } else {
                        outBf[(size_t)row * GH + col] = (bf16_t)vv;
                    }
                }
            }
        }
    }
}

// ---------------------------------------------------------------------------
// head: out[g][c] = relu(pooled[g]/cnt[g]) @ Wl + bl
__global__ __launch_bounds__(256) void k_head(const float* __restrict__ pooled,
                                              const float* __restrict__ cnt,
                                              const float* __restrict__ Wl,
                                              const float* __restrict__ bl,
                                              float* __restrict__ out) {
    __shared__ float red[256];
    int g = blockIdx.x, tid = threadIdx.x;
    float c = fmaxf(cnt[g], 1.0f);
    float a[GC];
#pragma unroll
    for (int j = 0; j < GC; j++) a[j] = 0.f;
    for (int f = tid; f < GH; f += 256) {
        float p = fmaxf(pooled[(size_t)g * GH + f] / c, 0.f);
#pragma unroll
        for (int j = 0; j < GC; j++) a[j] += p * Wl[f * GC + j];
    }
    for (int j = 0; j < GC; j++) {
        red[tid] = a[j];
        __syncthreads();
        for (int s = 128; s > 0; s >>= 1) {
            if (tid < s) red[tid] += red[tid + s];
            __syncthreads();
        }
        if (tid == 0) out[g * GC + j] = red[0] + bl[j];
        __syncthreads();
    }
}

// ---------------------------------------------------------------------------
extern "C" void kernel_launch(void* const* d_in, const int* in_sizes, int n_in,
                              void* d_out, int out_size, void* d_ws,
                              size_t ws_size, hipStream_t stream) {
    const float* x = (const float*)d_in[0];
    const int* ei = (const int*)d_in[1];
    const int* batch = (const int*)d_in[2];
    const float* Wr0 = (const float*)d_in[3];
    const float* br0 = (const float*)d_in[4];
    const float* Ws0 = (const float*)d_in[5];
    const float* Wr1 = (const float*)d_in[6];
    const float* br1 = (const float*)d_in[7];
    const float* Ws1 = (const float*)d_in[8];
    const float* Wl = (const float*)d_in[9];
    const float* bl = (const float*)d_in[10];
    float* out = (float*)d_out;
    (void)in_sizes; (void)n_in; (void)out_size; (void)ws_size;

    char* base = (char*)d_ws;
    size_t off = 0;
    auto take = [&](size_t bytes) {
        char* p = base + off;
        off = (off + bytes + 255) & ~(size_t)255;
        return p;
    };
    float* aggf   = (float*)take((size_t)GN * GD * 4);   // fp32 scatter target
    bf16_t* ab    = (bf16_t*)take((size_t)GN * GD * 2);  // agg in bf16
    bf16_t* xb    = (bf16_t*)take((size_t)GN * GD * 2);  // layer input bf16
    bf16_t* h1b   = (bf16_t*)take((size_t)GN * GD * 2);  // layer-0 output bf16
    bf16_t* wrb   = (bf16_t*)take((size_t)GD * GH * 2);  // Wr^T bf16
    bf16_t* wsb   = (bf16_t*)take((size_t)GD * GH * 2);  // Ws^T bf16
    float* pooled = (float*)take((size_t)GG * GH * 4);
    float* cnt    = (float*)take((size_t)GG * 4);

    const long long nx = (long long)GN * GD;
    const int cvtBlocks = (int)((nx + 255) / 256);
    const int wBlocks = (GD * GH + 255) / 256;
    const int aggBlocks = (GE + 7) / 8;  // one wave per edge, 8 waves/block
    dim3 gemmGrid(GH / 128, (GN + 127) / 128);

    // ---- layer 0 ----
    (void)hipMemsetAsync(aggf, 0, (size_t)GN * GD * 4, stream);
    k_f2bf<<<cvtBlocks, 256, 0, stream>>>(x, xb, nx);
    k_w2bft<<<wBlocks, 256, 0, stream>>>(Wr0, wrb);
    k_w2bft<<<wBlocks, 256, 0, stream>>>(Ws0, wsb);
    k_aggregate<float><<<aggBlocks, 256, 0, stream>>>(x, ei, aggf);
    k_f2bf<<<cvtBlocks, 256, 0, stream>>>(aggf, ab, nx);
    k_gcn_gemm<<<gemmGrid, 256, 0, stream>>>(ab, xb, wrb, wsb, br0, h1b,
                                             (float*)nullptr, (const int*)nullptr,
                                             GN, 0);

    // ---- layer 1 (fused mean-pool epilogue) ----
    (void)hipMemsetAsync(aggf, 0, (size_t)GN * GD * 4, stream);
    (void)hipMemsetAsync(pooled, 0, (size_t)GG * GH * 4, stream);
    (void)hipMemsetAsync(cnt, 0, (size_t)GG * 4, stream);
    k_aggregate<bf16_t><<<aggBlocks, 256, 0, stream>>>(h1b, ei, aggf);
    k_f2bf<<<cvtBlocks, 256, 0, stream>>>(aggf, ab, nx);
    k_w2bft<<<wBlocks, 256, 0, stream>>>(Wr1, wrb);
    k_w2bft<<<wBlocks, 256, 0, stream>>>(Ws1, wsb);
    k_count<<<(GN + 255) / 256, 256, 0, stream>>>(batch, cnt);
    k_gcn_gemm<<<gemmGrid, 256, 0, stream>>>(ab, h1b, wrb, wsb, br1,
                                             (bf16_t*)nullptr, pooled, batch,
                                             GN, 1);

    // ---- head ----
    k_head<<<GG, 256, 0, stream>>>(pooled, cnt, Wl, bl, out);
}
